// HierarchicalBlockAttention_52587579572813
// MI455X (gfx1250) — matrix operand order repaired and checked by reading, not compiled, not run on hardware
//
#include <hip/hip_runtime.h>

// ---------------------------------------------------------------------------
// HierarchicalBlockAttention for MI455X (gfx1250), wave32 + WMMA bf16 path.
//
// Pipeline:
//   1) proj_gemm<0>: Q = q @ Wq^T + bq   -> bf16 ws, layout [b,h,blk,row,d]
//      proj_gemm<0>: K = k @ Wk^T + bk   -> bf16 ws, layout [b,h,blk,row,d]
//      proj_gemm<1>: V = v @ Wv^T + bv   -> bf16 ws, layout [b,h,blk,d,row] (transposed)
//   2) attn_kernel : per (b,h,blk) 128x128 block attention, softmax fp32,
//                    all matmuls via v_wmma_f32_16x16x32_bf16
//                    -> bf16 ws, layout [token][h*64+d]
//   3) out_gemm    : out = attn @ Wo^T + bo -> fp32 d_out
//                    A tile staged into LDS by the Tensor Data Mover (TDM).
// ---------------------------------------------------------------------------

typedef __attribute__((ext_vector_type(4)))  __bf16 bf16x4;
typedef __attribute__((ext_vector_type(8)))  __bf16 bf16x8;
typedef __attribute__((ext_vector_type(16))) __bf16 bf16x16;
typedef __attribute__((ext_vector_type(8)))  float  f32x8;
typedef __attribute__((ext_vector_type(4)))  unsigned int u32x4;
typedef __attribute__((ext_vector_type(8)))  int    i32x8;
typedef __attribute__((ext_vector_type(4)))  int    i32x4;

#define D_MODEL 1024
#define NHEAD   16
#define D_K     64
#define D_BLOCK 128
#define BATCH   4
#define SEQ     4096
#define MTOT    (BATCH * SEQ)          // 16384
#define NBLK    (SEQ / D_BLOCK)        // 32

// LDS tile strides (bf16 elements). 72*2B = 144B = 16B-aligned & conflict-free.
#define ASTRIDE 72
#define PSTRIDE 136

static __device__ __forceinline__ bf16x16 cat16(bf16x8 lo, bf16x8 hi) {
    return __builtin_shufflevector(lo, hi, 0, 1, 2, 3, 4, 5, 6, 7,
                                           8, 9, 10, 11, 12, 13, 14, 15);
}

// Packed fp32x4 -> bf16x4 convert + single 8-byte store (v_cvt_pk_bf16_f32).
static __device__ __forceinline__ void cvt_store4(__bf16* p, float4 f) {
    bf16x4 o = { (__bf16)f.x, (__bf16)f.y, (__bf16)f.z, (__bf16)f.w };
    *(bf16x4*)p = o;
}

// A-matrix fragment (16x32 bf16), ISA 7.12.2 layout:
// lane<16 : row=lane,    elems 0..7 = K k0..k0+7,   elems 8..15 = K k0+16..k0+23
// lane>=16: row=lane-16, elems 0..7 = K k0+8..+15,  elems 8..15 = K k0+24..+31
static __device__ __forceinline__ bf16x16
frag_a_ld(const __bf16* base, int stride, int rbase, int k0, int lrow, int lhi) {
    const __bf16* p = base + (size_t)(rbase + lrow) * stride + k0 + lhi * 8;
    bf16x8 lo = *(const bf16x8*)p;
    bf16x8 hi = *(const bf16x8*)(p + 16);
    return cat16(lo, hi);
}

// B-matrix fragment (32x16 bf16): source stored as Bsrc[n][k] (row = output col).
// lane<16 : col=lane,    elems 0..7 = K k0..k0+7,   elems 8..15 = K k0+16..k0+23
// lane>=16: col=lane-16, elems 0..7 = K k0+8..+15,  elems 8..15 = K k0+24..+31
static __device__ __forceinline__ bf16x16
frag_b_ld(const __bf16* base, int stride, int nbase, int k0, int lrow, int lhi) {
    const __bf16* p = base + (size_t)(nbase + lrow) * stride + k0 + lhi * 8;
    bf16x8 lo = *(const bf16x8*)p;
    bf16x8 hi = *(const bf16x8*)(p + 16);
    return cat16(lo, hi);
}

static __device__ __forceinline__ f32x8
wmma_bf16(bf16x16 a, bf16x16 b, f32x8 c) {
    return __builtin_amdgcn_wmma_f32_16x16x32_bf16(
        false, a, false, b, (short)0, c, false, false);
}

// ---------------------------------------------------------------------------
// Tensor Data Mover: async-copy a 2D bf16 tile (tile_d0 elems x tile_d1 rows)
// from global to LDS, padding each 32-DWORD row with 4 DWORDs so LDS row
// stride becomes 72 bf16 (= ASTRIDE).  D# layout per CDNA5 ISA ch.8.
// This toolchain exposes the 6-operand builtin:
//   (u32x4 group0, i32x8 group1, i32x4 group2, i32x4 group3, i32x8 extra, i32 cpol)
// ---------------------------------------------------------------------------
static __device__ __forceinline__ void
tdm_load_tile_bf16(unsigned lds_byte_off, const void* gaddr,
                   unsigned tile_d0, unsigned tile_d1,
                   unsigned tensor_d0, unsigned tensor_d1,
                   unsigned row_stride_elems) {
    unsigned long long ga = (unsigned long long)(size_t)gaddr;
    u32x4 g0;
    g0[0] = 1u;                                          // count=1, user D#
    g0[1] = lds_byte_off;                                // lds_addr
    g0[2] = (unsigned)(ga & 0xFFFFFFFFu);                // global_addr[31:0]
    g0[3] = (unsigned)((ga >> 32) & 0x1FFFFFFu)          // global_addr[56:32]
            | (2u << 30);                                // type=2 (image)
    i32x8 g1;
    g1[0] = (int)((1u << 16)      // data_size = 1 (2 bytes)
                | (1u << 20)      // pad_enable
                | (4u << 22)      // pad_interval code 4 -> every 32 DWORDs
                | (3u << 25));    // pad_amount  code 3 -> 4 DWORDs (8 bf16)
    g1[1] = (int)((tensor_d0 & 0xFFFFu) << 16);                    // dim0 lo
    g1[2] = (int)(((tensor_d0 >> 16) & 0xFFFFu)
                | ((tensor_d1 & 0xFFFFu) << 16));                  // dim0 hi | dim1 lo
    g1[3] = (int)(((tensor_d1 >> 16) & 0xFFFFu)
                | ((tile_d0 & 0xFFFFu) << 16));                    // dim1 hi | tile_dim0
    g1[4] = (int)(tile_d1 & 0xFFFFu);                              // tile_dim1 (tile_dim2=0)
    g1[5] = (int)row_stride_elems;                                 // tensor_dim0_stride lo
    g1[6] = 0;
    g1[7] = 0;
    i32x4 gz4 = {0, 0, 0, 0};
    i32x8 gz8 = {0, 0, 0, 0, 0, 0, 0, 0};
    __builtin_amdgcn_tensor_load_to_lds(g0, g1, gz4, gz4, gz8, 0);
}

// ---------------------------------------------------------------------------
// Projection GEMM: C[m][n] = sum_k A[m][k] * W[n][k] + bias[n]
// A fp32 (M x 1024), W fp32 (1024 x 1024 row-major == B already in [n][k]).
// MODE 0: write bf16 dst at [b,h,blk,row,d]  (Q / K layout)
// MODE 1: write bf16 dst at [b,h,blk,d,row]  (V transposed layout)
// ---------------------------------------------------------------------------
template <int MODE>
__global__ __launch_bounds__(256)
void proj_gemm(const float* __restrict__ A, const float* __restrict__ W,
               const float* __restrict__ bias, __bf16* __restrict__ dst) {
    __shared__ __bf16 As[128 * ASTRIDE];
    __shared__ __bf16 Bs[128 * ASTRIDE];

    const int tid  = threadIdx.x;
    const int lane = tid & 31;
    const int wv   = tid >> 5;          // wave id 0..7 -> m sub-tile
    const int lrow = lane & 15;
    const int lhi  = lane >> 4;
    const int m0   = blockIdx.y * 128;
    const int n0   = blockIdx.x * 128;

    f32x8 acc[8];
    const f32x8 zero = {0.f, 0.f, 0.f, 0.f, 0.f, 0.f, 0.f, 0.f};
#pragma unroll
    for (int t = 0; t < 8; ++t) acc[t] = zero;

    for (int kb = 0; kb < D_MODEL / 64; ++kb) {
        // Stage 128x64 fp32 tiles of A and W into LDS as bf16 (packed cvt).
#pragma unroll
        for (int i = 0; i < 8; ++i) {
            int idx = tid + 256 * i;          // 0..2047
            int row = idx >> 4;
            int kq  = (idx & 15) << 2;
            float4 fa = *(const float4*)(A + (size_t)(m0 + row) * D_MODEL + kb * 64 + kq);
            cvt_store4(&As[row * ASTRIDE + kq], fa);
            float4 fb = *(const float4*)(W + (size_t)(n0 + row) * D_MODEL + kb * 64 + kq);
            cvt_store4(&Bs[row * ASTRIDE + kq], fb);
        }
        if (kb + 1 < D_MODEL / 64) {   // prefetch next tiles (global_prefetch_b8)
            __builtin_prefetch(&A[(size_t)(m0 + (tid & 127)) * D_MODEL + (kb + 1) * 64], 0, 0);
            __builtin_prefetch(&W[(size_t)(n0 + (tid & 127)) * D_MODEL + (kb + 1) * 64], 0, 0);
        }
        __syncthreads();

        bf16x16 a0 = frag_a_ld(As, ASTRIDE, 16 * wv, 0,  lrow, lhi);
        bf16x16 a1 = frag_a_ld(As, ASTRIDE, 16 * wv, 32, lrow, lhi);
#pragma unroll
        for (int t = 0; t < 8; ++t) {
            bf16x16 b0 = frag_b_ld(Bs, ASTRIDE, 16 * t, 0,  lrow, lhi);
            bf16x16 b1 = frag_b_ld(Bs, ASTRIDE, 16 * t, 32, lrow, lhi);
            acc[t] = wmma_bf16(a0, b0, acc[t]);
            acc[t] = wmma_bf16(a1, b1, acc[t]);
        }
        __syncthreads();
    }

    // Scatter results into per-head layout.
#pragma unroll
    for (int t = 0; t < 8; ++t) {
        int gcol = n0 + 16 * t + lrow;
        float bv = bias[gcol];
        int h = gcol >> 6;
        int d = gcol & 63;
#pragma unroll
        for (int v = 0; v < 8; ++v) {
            int grow = m0 + 16 * wv + 8 * lhi + v;
            float val = acc[t][v] + bv;
            int b   = grow >> 12;           // /4096
            int s   = grow & 4095;
            int blk = s >> 7;               // /128
            int r   = s & 127;
            size_t base = (size_t)((b * NHEAD + h) * NBLK + blk) * (D_BLOCK * D_K);
            if (MODE == 0)
                dst[base + (size_t)r * D_K + d] = (__bf16)val;      // [row][d]
            else
                dst[base + (size_t)d * D_BLOCK + r] = (__bf16)val;  // [d][row]
        }
    }
}

// ---------------------------------------------------------------------------
// Block attention: one workgroup per (b,h,blk). 8 waves, wave w owns rows
// 16w..16w+15 of the 128-row block.
// ---------------------------------------------------------------------------
__global__ __launch_bounds__(256)
void attn_kernel(const __bf16* __restrict__ Q, const __bf16* __restrict__ K,
                 const __bf16* __restrict__ Vt, __bf16* __restrict__ O) {
    __shared__ __bf16 Plds[128 * PSTRIDE];

    const int bid  = blockIdx.x;            // (b*16+h)*32+blk
    const int blk  = bid & 31;
    const int h    = (bid >> 5) & 15;
    const int b    = bid >> 9;
    const size_t base = (size_t)bid * (D_BLOCK * D_K);
    const __bf16* Qb = Q + base;            // [128][64]
    const __bf16* Kb = K + base;            // [128][64]
    const __bf16* Vb = Vt + base;           // [64][128] (transposed V)

    const int tid  = threadIdx.x;
    const int lane = tid & 31;
    const int wv   = tid >> 5;
    const int lrow = lane & 15;
    const int lhi  = lane >> 4;

    // ---- S = (Q * scale) @ K^T : 16x128 per wave ----
    bf16x16 q0 = frag_a_ld(Qb, D_K, 16 * wv, 0,  lrow, lhi);
    bf16x16 q1 = frag_a_ld(Qb, D_K, 16 * wv, 32, lrow, lhi);

    f32x8 sc[8];
    const f32x8 zero = {0.f, 0.f, 0.f, 0.f, 0.f, 0.f, 0.f, 0.f};
#pragma unroll
    for (int t = 0; t < 8; ++t) sc[t] = zero;
#pragma unroll
    for (int t = 0; t < 8; ++t) {
        bf16x16 k0 = frag_b_ld(Kb, D_K, 16 * t, 0,  lrow, lhi);
        bf16x16 k1 = frag_b_ld(Kb, D_K, 16 * t, 32, lrow, lhi);
        sc[t] = wmma_bf16(q0, k0, sc[t]);
        sc[t] = wmma_bf16(q1, k1, sc[t]);
    }
    const float scale = 0.125f;             // 1/sqrt(64)
#pragma unroll
    for (int t = 0; t < 8; ++t)
#pragma unroll
        for (int v = 0; v < 8; ++v) sc[t][v] *= scale;

    // ---- softmax over each row (C-fragment: VGPR v -> row 16w+8*lhi+v,
    //      lane lrow -> col 16t+lrow; row lives in a 16-lane half-wave) ----
#pragma unroll
    for (int v = 0; v < 8; ++v) {
        float m = -3.4e38f;
#pragma unroll
        for (int t = 0; t < 8; ++t) m = fmaxf(m, sc[t][v]);
        for (int mk = 1; mk < 16; mk <<= 1) m = fmaxf(m, __shfl_xor(m, mk, 32));
        float sum = 0.f;
#pragma unroll
        for (int t = 0; t < 8; ++t) {
            float e = __expf(sc[t][v] - m);
            sc[t][v] = e;
            sum += e;
        }
        for (int mk = 1; mk < 16; mk <<= 1) sum += __shfl_xor(sum, mk, 32);
        float inv = 1.f / (sum + 1e-10f);
        int r = 16 * wv + 8 * lhi + v;
#pragma unroll
        for (int t = 0; t < 8; ++t)
            Plds[r * PSTRIDE + 16 * t + lrow] = (__bf16)(sc[t][v] * inv);
    }
    __syncthreads();

    // ---- O = P @ V : 16x64 per wave, contraction K=128 ----
#pragma unroll
    for (int tv = 0; tv < 4; ++tv) {
        f32x8 ao = zero;
#pragma unroll
        for (int kc = 0; kc < 4; ++kc) {
            bf16x16 pa = frag_a_ld(Plds, PSTRIDE, 16 * wv, 32 * kc, lrow, lhi);
            bf16x16 vb = frag_b_ld(Vb, D_BLOCK, 16 * tv, 32 * kc, lrow, lhi);
            ao = wmma_bf16(pa, vb, ao);
        }
        int d = 16 * tv + lrow;
#pragma unroll
        for (int v = 0; v < 8; ++v) {
            int r = 16 * wv + 8 * lhi + v;
            size_t tok = (size_t)(b * SEQ + blk * D_BLOCK + r);
            O[tok * D_MODEL + h * D_K + d] = (__bf16)ao[v];
        }
    }
}

// ---------------------------------------------------------------------------
// Output GEMM: out[m][n] = sum_k attn_bf16[m][k] * Wo[n][k] + bo[n]  (fp32 out)
// A tile (already bf16 in global) is staged to LDS by the Tensor Data Mover.
// ---------------------------------------------------------------------------
__global__ __launch_bounds__(256)
void out_gemm(const __bf16* __restrict__ A, const float* __restrict__ W,
              const float* __restrict__ bias, float* __restrict__ out) {
    __shared__ __bf16 As[128 * ASTRIDE];
    __shared__ __bf16 Bs[128 * ASTRIDE];

    const int tid  = threadIdx.x;
    const int lane = tid & 31;
    const int wv   = tid >> 5;
    const int lrow = lane & 15;
    const int lhi  = lane >> 4;
    const int m0   = blockIdx.y * 128;
    const int n0   = blockIdx.x * 128;

    const unsigned as_lds = (unsigned)(size_t)(&As[0]);   // LDS byte offset

    f32x8 acc[8];
    const f32x8 zero = {0.f, 0.f, 0.f, 0.f, 0.f, 0.f, 0.f, 0.f};
#pragma unroll
    for (int t = 0; t < 8; ++t) acc[t] = zero;

    for (int kb = 0; kb < D_MODEL / 64; ++kb) {
        // A tile: bf16 128x64 via TDM async DMA (pads rows 64 -> 72 bf16).
        if (wv == 0) {
            tdm_load_tile_bf16(as_lds,
                               A + (size_t)m0 * D_MODEL + kb * 64,
                               /*tile_d0=*/64, /*tile_d1=*/128,
                               /*tensor_d0=*/D_MODEL, /*tensor_d1=*/MTOT,
                               /*row_stride=*/D_MODEL);
            __builtin_amdgcn_s_wait_tensorcnt(0);
        }
        // W tile: fp32 -> bf16 on the fly (packed cvt + b64 LDS stores).
#pragma unroll
        for (int i = 0; i < 8; ++i) {
            int idx = tid + 256 * i;
            int row = idx >> 4;
            int kq  = (idx & 15) << 2;
            float4 fb = *(const float4*)(W + (size_t)(n0 + row) * D_MODEL + kb * 64 + kq);
            cvt_store4(&Bs[row * ASTRIDE + kq], fb);
        }
        if (kb + 1 < D_MODEL / 64) {
            __builtin_prefetch(&A[(size_t)(m0 + (tid & 127)) * D_MODEL + (kb + 1) * 64], 0, 0);
            __builtin_prefetch(&W[(size_t)(n0 + (tid & 127)) * D_MODEL + (kb + 1) * 64], 0, 0);
        }
        __syncthreads();

        bf16x16 a0 = frag_a_ld(As, ASTRIDE, 16 * wv, 0,  lrow, lhi);
        bf16x16 a1 = frag_a_ld(As, ASTRIDE, 16 * wv, 32, lrow, lhi);
#pragma unroll
        for (int t = 0; t < 8; ++t) {
            bf16x16 b0 = frag_b_ld(Bs, ASTRIDE, 16 * t, 0,  lrow, lhi);
            bf16x16 b1 = frag_b_ld(Bs, ASTRIDE, 16 * t, 32, lrow, lhi);
            acc[t] = wmma_bf16(a0, b0, acc[t]);
            acc[t] = wmma_bf16(a1, b1, acc[t]);
        }
        __syncthreads();
    }

#pragma unroll
    for (int t = 0; t < 8; ++t) {
        int gcol = n0 + 16 * t + lrow;
        float bv = bias[gcol];
#pragma unroll
        for (int v = 0; v < 8; ++v) {
            int grow = m0 + 16 * wv + 8 * lhi + v;
            out[(size_t)grow * D_MODEL + gcol] = acc[t][v] + bv;
        }
    }
}

// ---------------------------------------------------------------------------
// Host launcher
// ---------------------------------------------------------------------------
extern "C" void kernel_launch(void* const* d_in, const int* in_sizes, int n_in,
                              void* d_out, int out_size, void* d_ws, size_t ws_size,
                              hipStream_t stream) {
    (void)in_sizes; (void)n_in; (void)out_size; (void)ws_size;

    const float* q  = (const float*)d_in[0];
    const float* k  = (const float*)d_in[1];
    const float* v  = (const float*)d_in[2];
    const float* Wq = (const float*)d_in[3];
    const float* bq = (const float*)d_in[4];
    const float* Wk = (const float*)d_in[5];
    const float* bk = (const float*)d_in[6];
    const float* Wv = (const float*)d_in[7];
    const float* bv = (const float*)d_in[8];
    const float* Wo = (const float*)d_in[9];
    const float* bo = (const float*)d_in[10];
    float* out = (float*)d_out;

    const size_t TEN = (size_t)MTOT * D_MODEL;   // 16,777,216 elements
    __bf16* ws  = (__bf16*)d_ws;
    __bf16* Qw  = ws;
    __bf16* Kw  = ws + TEN;
    __bf16* Vw  = ws + 2 * TEN;
    __bf16* Aw  = ws + 3 * TEN;

    dim3 gblk(256);
    dim3 ggrid(D_MODEL / 128, MTOT / 128);       // 8 x 128

    proj_gemm<0><<<ggrid, gblk, 0, stream>>>(q, Wq, bq, Qw);
    proj_gemm<0><<<ggrid, gblk, 0, stream>>>(k, Wk, bk, Kw);
    proj_gemm<1><<<ggrid, gblk, 0, stream>>>(v, Wv, bv, Vw);

    attn_kernel<<<BATCH * NHEAD * NBLK, gblk, 0, stream>>>(Qw, Kw, Vw, Aw);

    out_gemm<<<ggrid, gblk, 0, stream>>>(Aw, Wo, bo, out);
}
